// Superpoint_sample_910533067698
// MI455X (gfx1250) — compile-verified, matrix operand order checked by reading, not run on hardware
//
#include <hip/hip_runtime.h>
#include <hip/hip_bf16.h>

typedef __attribute__((ext_vector_type(16))) __bf16 v16bf;
typedef __attribute__((ext_vector_type(8)))  __bf16 v8bf;
typedef __attribute__((ext_vector_type(4)))  __bf16 v4bf;
typedef __attribute__((ext_vector_type(8)))  float  v8f;

#define C_ATT 256
#define NPTS  2048
#define BATCH 4
#define HEADS 8
#define DH    32
#define SQ    16      // query points for grouping
#define KS    33      // 1 + NSAMPLE
#define SK    (SQ*KS) // 528

// K-index mapping for 16-bit WMMA A/B fragments (ISA 7.12.2):
// lanes 0-15 hold K 0..7 and 16..23; lanes 16-31 hold K 8..15 and 24..31.
__device__ __forceinline__ int kmap(int e, int half) {
  return (e & 7) + ((e >> 3) << 4) + half * 8;
}

// Build a 16-element bf16 fragment from LDS: elements {p[0..7], p[16..23]}.
// p must be 16B aligned -> two ds_load_b128, merged by shufflevector.
__device__ __forceinline__ v16bf frag16(const __bf16* p) {
  const v8bf* pv = (const v8bf*)p;
  v8bf lo = pv[0];
  v8bf hi = pv[2];
  return __builtin_shufflevector(lo, hi, 0, 1, 2, 3, 4, 5, 6, 7,
                                 8, 9, 10, 11, 12, 13, 14, 15);
}

__device__ __forceinline__ v4bf cvt4(float4 f) {
  v4bf h;
  h[0] = (__bf16)f.x; h[1] = (__bf16)f.y; h[2] = (__bf16)f.z; h[3] = (__bf16)f.w;
  return h;
}

// ---------------------------------------------------------------------------
// Tiled bf16 WMMA GEMM: C[M,N] = A[M,K] * B[K,N], f32 in/out, bf16 compute.
// Block tile 128x64, 8 waves (4x2), each wave 32x32 (2x2 WMMA frags), K step 32.
// B tile is stored transposed in LDS so both operand fragments are contiguous.
// ---------------------------------------------------------------------------
__global__ __launch_bounds__(256) void gemm_bf16_wmma(
    const float* __restrict__ A, const float* __restrict__ Bw,
    float* __restrict__ C, int M, int N, int K) {
  __shared__ __bf16 sA[128][40];   // 128x32 (+8 pad) = 10KB
  __shared__ __bf16 sBt[64][40];   // 64n x 32k transposed (+8 pad) = 5KB

  const int tid  = threadIdx.x;
  const int lane = tid & 31;
  const int wid  = tid >> 5;
  const int wm   = wid & 3;
  const int wn   = wid >> 2;
  const int half = lane >> 4;
  const int l16  = lane & 15;

  const int m0 = blockIdx.x * 128;
  const int n0 = blockIdx.y * 64;

  v8f acc[2][2] = {};

  for (int k0 = 0; k0 < K; k0 += 32) {
    // stage A: 128x32, float4 loads, v4bf stores (4 iters)
    for (int i = tid; i < 128 * 8; i += 256) {
      int r = i >> 3, c = (i & 7) << 2;
      float4 f = *(const float4*)&A[(size_t)(m0 + r) * K + k0 + c];
      *(v4bf*)&sA[r][c] = cvt4(f);
    }
    // stage B transposed: lane-consecutive n => coalesced global, v4bf stores
    for (int i = tid; i < 64 * 8; i += 256) {
      int n = i & 63, kq = (i >> 6) << 2;
      v4bf h;
      #pragma unroll
      for (int j = 0; j < 4; ++j)
        h[j] = (__bf16)Bw[(size_t)(k0 + kq + j) * N + n0 + n];
      *(v4bf*)&sBt[n][kq] = h;
    }
    __syncthreads();

    #pragma unroll
    for (int mi = 0; mi < 2; ++mi) {
      const v16bf a = frag16(&sA[wm * 32 + mi * 16 + l16][half * 8]);
      #pragma unroll
      for (int ni = 0; ni < 2; ++ni) {
        const v16bf b = frag16(&sBt[wn * 32 + ni * 16 + l16][half * 8]);
        acc[mi][ni] = __builtin_amdgcn_wmma_f32_16x16x32_bf16(
            false, a, false, b, (short)0, acc[mi][ni], false, false);
      }
    }
    __syncthreads();
  }

  #pragma unroll
  for (int mi = 0; mi < 2; ++mi)
    #pragma unroll
    for (int ni = 0; ni < 2; ++ni)
      #pragma unroll
      for (int r = 0; r < 8; ++r) {
        int row = m0 + wm * 32 + mi * 16 + r + half * 8;
        int col = n0 + wn * 32 + ni * 16 + l16;
        C[(size_t)row * N + col] = acc[mi][ni][r];
      }
}

// ---------------------------------------------------------------------------
// Flash attention: block = (64 query rows, one (b,h)); 4 waves x 16 rows each.
// Online softmax; QK^T and PV via v_wmma_f32_16x16x32_bf16.
// V is stored transposed in LDS so the PV B-fragment is key-contiguous.
// ---------------------------------------------------------------------------
__global__ __launch_bounds__(128) void attn_wmma(
    const float* __restrict__ Q, const float* __restrict__ Kb,
    const float* __restrict__ Vb, float* __restrict__ O) {
  __shared__ __bf16 sK[64][40];       // keys x dh (+pad)     5KB
  __shared__ __bf16 sVt[32][72];      // dh x keys (+pad)     4.5KB
  __shared__ float  sS[4][16][64];    // scores               16KB
  __shared__ __bf16 sP[4][16][72];    // probabilities (+pad) 9KB
  __shared__ float  sScale[4][16];
  __shared__ float  sLinv[4][16];

  const int tid  = threadIdx.x;
  const int lane = tid & 31;
  const int wid  = tid >> 5;   // 0..3
  const int half = lane >> 4;
  const int l16  = lane & 15;

  const int bh = blockIdx.y;
  const int b  = bh / HEADS;
  const int h  = bh % HEADS;
  const int q0 = blockIdx.x * 64;

  const float scale = 0.17677669529663687f;  // 1/sqrt(32), folded into Q

  v16bf qf;
  {
    const int row = q0 + wid * 16 + l16;
    const float* qp = Q + ((size_t)(b * NPTS + row) * C_ATT) + h * DH;
    #pragma unroll
    for (int e = 0; e < 16; ++e) qf[e] = (__bf16)(qp[kmap(e, half)] * scale);
  }

  float m_row = -3.0e38f, l_row = 0.0f;  // valid in lanes 0..15
  v8f o_acc[2] = {};

  for (int kt = 0; kt < NPTS; kt += 64) {
    // K tile: float4 along dh, coalesced
    for (int i = tid; i < 64 * 8; i += 128) {
      int r = i >> 3, c = (i & 7) << 2;
      float4 f = *(const float4*)&Kb[((size_t)(b * NPTS + kt + r) * C_ATT) + h * DH + c];
      *(v4bf*)&sK[r][c] = cvt4(f);
    }
    // V tile transposed: lane-consecutive d => coalesced global reads
    for (int i = tid; i < 32 * 16; i += 128) {
      int d = i & 31, kq = (i >> 5) << 2;
      v4bf hh;
      #pragma unroll
      for (int j = 0; j < 4; ++j)
        hh[j] = (__bf16)Vb[((size_t)(b * NPTS + kt + kq + j) * C_ATT) + h * DH + d];
      *(v4bf*)&sVt[d][kq] = hh;
    }
    __syncthreads();

    #pragma unroll
    for (int nt = 0; nt < 4; ++nt) {
      const v16bf kf = frag16(&sK[nt * 16 + l16][half * 8]);
      v8f s = {};
      s = __builtin_amdgcn_wmma_f32_16x16x32_bf16(
          false, qf, false, kf, (short)0, s, false, false);
      #pragma unroll
      for (int r = 0; r < 8; ++r)
        sS[wid][r + half * 8][nt * 16 + l16] = s[r];
    }
    __syncthreads();

    if (half == 0) {  // lanes 0..15: one row each
      float mt = -3.0e38f;
      #pragma unroll 8
      for (int j = 0; j < 64; ++j) mt = fmaxf(mt, sS[wid][l16][j]);
      float mn   = fmaxf(m_row, mt);
      float corr = __expf(m_row - mn);
      float ls = 0.0f;
      #pragma unroll 8
      for (int j = 0; j < 64; ++j) {
        float p = __expf(sS[wid][l16][j] - mn);
        sP[wid][l16][j] = (__bf16)p;
        ls += p;
      }
      l_row = l_row * corr + ls;
      m_row = mn;
      sScale[wid][l16] = corr;
    }
    __syncthreads();

    #pragma unroll
    for (int r = 0; r < 8; ++r) {
      float corr = sScale[wid][r + half * 8];
      o_acc[0][r] *= corr;
      o_acc[1][r] *= corr;
    }

    #pragma unroll
    for (int kc = 0; kc < 2; ++kc) {
      const v16bf pf = frag16(&sP[wid][l16][kc * 32 + half * 8]);
      #pragma unroll
      for (int dt = 0; dt < 2; ++dt) {
        const v16bf vf = frag16(&sVt[dt * 16 + l16][kc * 32 + half * 8]);
        o_acc[dt] = __builtin_amdgcn_wmma_f32_16x16x32_bf16(
            false, pf, false, vf, (short)0, o_acc[dt], false, false);
      }
    }
    __syncthreads();
  }

  if (half == 0) sLinv[wid][l16] = 1.0f / l_row;
  __syncthreads();

  #pragma unroll
  for (int dt = 0; dt < 2; ++dt)
    #pragma unroll
    for (int r = 0; r < 8; ++r) {
      int row = q0 + wid * 16 + r + half * 8;
      O[((size_t)(b * NPTS + row) * C_ATT) + h * DH + dt * 16 + l16] =
          o_acc[dt][r] * sLinv[wid][r + half * 8];
    }
}

// ---------------------------------------------------------------------------
// Residual + LayerNorm over C=256; one block per token row.
// ---------------------------------------------------------------------------
__global__ __launch_bounds__(256) void add_ln(
    const float* __restrict__ x, const float* __restrict__ o,
    const float* __restrict__ g, const float* __restrict__ bta,
    float* __restrict__ y) {
  __shared__ float red[256];
  const int row = blockIdx.x, c = threadIdx.x;
  float v = x[(size_t)row * C_ATT + c] + o[(size_t)row * C_ATT + c];
  red[c] = v; __syncthreads();
  for (int s = 128; s > 0; s >>= 1) { if (c < s) red[c] += red[c + s]; __syncthreads(); }
  float mean = red[0] * (1.0f / C_ATT); __syncthreads();
  float d = v - mean;
  red[c] = d * d; __syncthreads();
  for (int s = 128; s > 0; s >>= 1) { if (c < s) red[c] += red[c + s]; __syncthreads(); }
  float var = red[0] * (1.0f / C_ATT);
  y[(size_t)row * C_ATT + c] = d * __frsqrt_rn(var + 1e-5f) * g[c] + bta[c];
}

// ---------------------------------------------------------------------------
// Layout shuffles
// ---------------------------------------------------------------------------
__global__ void bcn_to_bnc(const float* __restrict__ in, float* __restrict__ out) {
  int i = blockIdx.x * blockDim.x + threadIdx.x;
  if (i >= BATCH * C_ATT * NPTS) return;
  int c = i % C_ATT, n = (i / C_ATT) % NPTS, b = i / (C_ATT * NPTS);
  out[i] = in[((size_t)b * C_ATT + c) * NPTS + n];
}
__global__ void bnc_to_bcn(const float* __restrict__ in, float* __restrict__ out) {
  int i = blockIdx.x * blockDim.x + threadIdx.x;
  if (i >= BATCH * C_ATT * NPTS) return;
  int n = i % NPTS, c = (i / NPTS) % C_ATT, b = i / (NPTS * C_ATT);
  out[i] = in[((size_t)b * NPTS + n) * C_ATT + c];
}

// ---------------------------------------------------------------------------
// 1x1 conv / linear over (B, Cin, S) -> (B, Cout, S)
// ---------------------------------------------------------------------------
__global__ void conv1x1(const float* __restrict__ in, const float* __restrict__ w,
                        const float* __restrict__ bias, float* __restrict__ out,
                        int Bc, int Cin, int Cout, int S) {
  int i = blockIdx.x * blockDim.x + threadIdx.x;
  if (i >= Bc * Cout * S) return;
  int s = i % S, o = (i / S) % Cout, b = i / (S * Cout);
  const float* ip = in + (size_t)b * Cin * S + s;
  const float* wp = w + (size_t)o * Cin;
  float acc = bias ? bias[o] : 0.0f;
  for (int c = 0; c < Cin; ++c) acc += wp[c] * ip[(size_t)c * S];
  out[i] = acc;
}

// Batch statistics per channel over (B, C, S): mean + 1/sqrt(var+eps)
__global__ __launch_bounds__(256) void chan_stats(
    const float* __restrict__ x, float* __restrict__ mean,
    float* __restrict__ rstd, int Bc, int Cc, int S) {
  __shared__ float rs[256], rq[256];
  const int c = blockIdx.x;
  float s0 = 0.0f, s1 = 0.0f;
  for (int b = 0; b < Bc; ++b) {
    const float* p = x + ((size_t)b * Cc + c) * S;
    for (int i = threadIdx.x; i < S; i += 256) { float v = p[i]; s0 += v; s1 += v * v; }
  }
  rs[threadIdx.x] = s0; rq[threadIdx.x] = s1; __syncthreads();
  for (int s = 128; s > 0; s >>= 1) {
    if (threadIdx.x < s) { rs[threadIdx.x] += rs[threadIdx.x + s]; rq[threadIdx.x] += rq[threadIdx.x + s]; }
    __syncthreads();
  }
  if (threadIdx.x == 0) {
    float n = (float)(Bc * S);
    float m = rs[0] / n;
    mean[c] = m;
    rstd[c] = __frsqrt_rn(rq[0] / n - m * m + 1e-5f);
  }
}

// In-place BN + activation (slope=0 -> relu, slope=0.2 -> leaky relu)
__global__ void bn_act(float* __restrict__ x, const float* __restrict__ mean,
                       const float* __restrict__ rstd, const float* __restrict__ g,
                       const float* __restrict__ bta, int Cc, int S, int total,
                       float slope) {
  int i = blockIdx.x * blockDim.x + threadIdx.x;
  if (i >= total) return;
  int c = (i / S) % Cc;
  float v = (x[i] - mean[c]) * rstd[c] * g[c] + bta[c];
  x[i] = v > 0.0f ? v : slope * v;
}

// Argmax over N per (b, channel)  (== top_k indices[...,0])
__global__ __launch_bounds__(256) void argmax_n(
    const float* __restrict__ h, int* __restrict__ idx, int S) {
  __shared__ float bv[256]; __shared__ int bi[256];
  const float* p = h + (size_t)blockIdx.x * S;
  float best = -3.0e38f; int besti = 0;
  for (int i = threadIdx.x; i < S; i += 256) {
    float v = p[i];
    if (v > best || (v == best && i < besti)) { best = v; besti = i; }
  }
  bv[threadIdx.x] = best; bi[threadIdx.x] = besti; __syncthreads();
  for (int s = 128; s > 0; s >>= 1) {
    if (threadIdx.x < s) {
      if (bv[threadIdx.x + s] > bv[threadIdx.x] ||
          (bv[threadIdx.x + s] == bv[threadIdx.x] && bi[threadIdx.x + s] < bi[threadIdx.x])) {
        bv[threadIdx.x] = bv[threadIdx.x + s]; bi[threadIdx.x] = bi[threadIdx.x + s];
      }
    }
    __syncthreads();
  }
  if (threadIdx.x == 0) idx[blockIdx.x] = bi[0];
}

// Ball query: first 32 (smallest index) points with sqd <= R^2, padded with gi[0]
__global__ void ball_query(const float* __restrict__ pts, const int* __restrict__ idx,
                           int* __restrict__ gi) {
  if (threadIdx.x != 0) return;
  const int bs = blockIdx.x;          // b*16 + s
  const int b  = bs >> 4;
  const int qi = idx[bs];
  const float* px = pts + (size_t)b * 3 * NPTS;
  float qx = px[qi], qy = px[NPTS + qi], qz = px[2 * NPTS + qi];
  int cnt = 0;
  int* gp = gi + bs * 32;
  for (int n = 0; n < NPTS && cnt < 32; ++n) {
    float dx = px[n] - qx, dy = px[NPTS + n] - qy, dz = px[2 * NPTS + n] - qz;
    if (dx * dx + dy * dy + dz * dz <= 0.09f) gp[cnt++] = n;
  }
  int pad = cnt > 0 ? gp[0] : qi;
  for (; cnt < 32; ++cnt) gp[cnt] = pad;
}

// Build grouped coords g: (B, 3, S*K) with k=0 the query point
__global__ void build_group(const float* __restrict__ pts, const int* __restrict__ idx,
                            const int* __restrict__ gi, float* __restrict__ g) {
  int i = blockIdx.x * blockDim.x + threadIdx.x;
  if (i >= BATCH * 3 * SK) return;
  int sk = i % SK, c = (i / SK) % 3, b = i / (SK * 3);
  int s = sk / KS, kk = sk % KS;
  int pidx = (kk == 0) ? idx[b * SQ + s] : gi[(b * SQ + s) * 32 + kk - 1];
  g[i] = pts[((size_t)b * 3 + c) * NPTS + pidx];
}

// Final max over k: (B,512,S*K) -> (B,512,16)
__global__ void max_over_k(const float* __restrict__ c3, float* __restrict__ out) {
  int i = blockIdx.x * blockDim.x + threadIdx.x;
  if (i >= BATCH * 512 * SQ) return;
  int s = i % SQ, o = (i / SQ) % 512, b = i / (SQ * 512);
  const float* p = c3 + ((size_t)b * 512 + o) * SK + s * KS;
  float m = p[0];
  for (int kk = 1; kk < KS; ++kk) m = fmaxf(m, p[kk]);
  out[i] = m;
}

// ---------------------------------------------------------------------------
extern "C" void kernel_launch(void* const* d_in, const int* in_sizes, int n_in,
                              void* d_out, int out_size, void* d_ws, size_t ws_size,
                              hipStream_t stream) {
  (void)in_sizes; (void)n_in; (void)out_size; (void)ws_size;
  const float* hf    = (const float*)d_in[0];
  const float* pts   = (const float*)d_in[1];
  const float* Wq    = (const float*)d_in[2];
  const float* Wk    = (const float*)d_in[3];
  const float* Wv    = (const float*)d_in[4];
  const float* Wo    = (const float*)d_in[5];
  const float* ln_g  = (const float*)d_in[6];
  const float* ln_b  = (const float*)d_in[7];
  const float* fg_w1 = (const float*)d_in[8];
  const float* fg_b1 = (const float*)d_in[9];
  const float* fg_w2 = (const float*)d_in[10];
  const float* fg_b2 = (const float*)d_in[11];
  const float* fg_w3 = (const float*)d_in[12];
  const float* fg_b3 = (const float*)d_in[13];
  const float* fg_g1 = (const float*)d_in[14];
  const float* fg_be1= (const float*)d_in[15];
  const float* fg_g2 = (const float*)d_in[16];
  const float* fg_be2= (const float*)d_in[17];
  const float* fg_g3 = (const float*)d_in[18];
  const float* fg_be3= (const float*)d_in[19];
  const float* cw1   = (const float*)d_in[20];
  const float* cw2   = (const float*)d_in[21];
  const float* cw3   = (const float*)d_in[22];
  const float* cg1   = (const float*)d_in[23];
  const float* cb1   = (const float*)d_in[24];
  const float* cg2   = (const float*)d_in[25];
  const float* cb2   = (const float*)d_in[26];
  const float* cg3   = (const float*)d_in[27];
  const float* cb3   = (const float*)d_in[28];
  float* out = (float*)d_out;

  const int M = BATCH * NPTS;               // 8192 tokens
  char* wp = (char*)d_ws;
  auto carve = [&](size_t bytes) -> void* {
    void* r = (void*)wp; wp += (bytes + 255) & ~(size_t)255; return r;
  };
  const size_t tok = (size_t)M * C_ATT * sizeof(float);   // 8 MB
  float* x     = (float*)carve(tok);
  float* x2    = (float*)carve(tok);
  float* qb    = (float*)carve(tok);
  float* kb    = (float*)carve(tok);
  float* vb    = (float*)carve(tok);
  float* ao    = (float*)carve(tok);
  float* oproj = (float*)carve(tok);
  float* feats = (float*)carve(tok);
  float* h1    = (float*)carve((size_t)BATCH * 64 * NPTS * sizeof(float));
  float* h2    = (float*)carve((size_t)BATCH * 32 * NPTS * sizeof(float));
  float* h3    = (float*)carve((size_t)BATCH * 16 * NPTS * sizeof(float));
  float* mean  = (float*)carve(512 * sizeof(float));
  float* rstd  = (float*)carve(512 * sizeof(float));
  int*   idx   = (int*)carve(BATCH * SQ * sizeof(int));
  int*   gi    = (int*)carve(BATCH * SQ * 32 * sizeof(int));
  float* gbuf  = (float*)carve((size_t)BATCH * 3 * SK * sizeof(float));
  float* c1    = (float*)carve((size_t)BATCH * 64 * SK * sizeof(float));
  float* c2    = (float*)carve((size_t)BATCH * 256 * SK * sizeof(float));
  float* c3    = (float*)carve((size_t)BATCH * 512 * SK * sizeof(float));

  auto blocks = [](int total) { return (total + 255) / 256; };

  // (B,C,N) -> tokens (B*N, C)
  bcn_to_bnc<<<blocks(M * C_ATT), 256, 0, stream>>>(hf, x);

  const dim3 ggrid(M / 128, C_ATT / 64);
  const dim3 agrid(NPTS / 64, BATCH * HEADS);

  for (int l = 0; l < 2; ++l) {
    const size_t wo = (size_t)l * C_ATT * C_ATT;
    gemm_bf16_wmma<<<ggrid, 256, 0, stream>>>(x, Wq + wo, qb, M, C_ATT, C_ATT);
    gemm_bf16_wmma<<<ggrid, 256, 0, stream>>>(x, Wk + wo, kb, M, C_ATT, C_ATT);
    gemm_bf16_wmma<<<ggrid, 256, 0, stream>>>(x, Wv + wo, vb, M, C_ATT, C_ATT);
    attn_wmma<<<agrid, 128, 0, stream>>>(qb, kb, vb, ao);
    gemm_bf16_wmma<<<ggrid, 256, 0, stream>>>(ao, Wo + wo, oproj, M, C_ATT, C_ATT);
    add_ln<<<M, 256, 0, stream>>>(x, oproj, ln_g + l * C_ATT, ln_b + l * C_ATT, x2);
    float* t = x; x = x2; x2 = t;
  }

  // feats (B, C, N)
  bnc_to_bcn<<<blocks(M * C_ATT), 256, 0, stream>>>(x, feats);

  // fg MLP with batch-norm (true batch statistics) + relu
  conv1x1<<<blocks(BATCH * 64 * NPTS), 256, 0, stream>>>(feats, fg_w1, fg_b1, h1, BATCH, 256, 64, NPTS);
  chan_stats<<<64, 256, 0, stream>>>(h1, mean, rstd, BATCH, 64, NPTS);
  bn_act<<<blocks(BATCH * 64 * NPTS), 256, 0, stream>>>(h1, mean, rstd, fg_g1, fg_be1, 64, NPTS, BATCH * 64 * NPTS, 0.0f);
  conv1x1<<<blocks(BATCH * 32 * NPTS), 256, 0, stream>>>(h1, fg_w2, fg_b2, h2, BATCH, 64, 32, NPTS);
  chan_stats<<<32, 256, 0, stream>>>(h2, mean, rstd, BATCH, 32, NPTS);
  bn_act<<<blocks(BATCH * 32 * NPTS), 256, 0, stream>>>(h2, mean, rstd, fg_g2, fg_be2, 32, NPTS, BATCH * 32 * NPTS, 0.0f);
  conv1x1<<<blocks(BATCH * 16 * NPTS), 256, 0, stream>>>(h2, fg_w3, fg_b3, h3, BATCH, 32, 16, NPTS);
  chan_stats<<<16, 256, 0, stream>>>(h3, mean, rstd, BATCH, 16, NPTS);
  bn_act<<<blocks(BATCH * 16 * NPTS), 256, 0, stream>>>(h3, mean, rstd, fg_g3, fg_be3, 16, NPTS, BATCH * 16 * NPTS, 0.0f);

  // query selection + ball query + grouping
  argmax_n<<<BATCH * SQ, 256, 0, stream>>>(h3, idx, NPTS);
  ball_query<<<BATCH * SQ, 32, 0, stream>>>(pts, idx, gi);
  build_group<<<blocks(BATCH * 3 * SK), 256, 0, stream>>>(pts, idx, gi, gbuf);

  // grouped conv stack with BN + leaky relu (0.2)
  conv1x1<<<blocks(BATCH * 64 * SK), 256, 0, stream>>>(gbuf, cw1, nullptr, c1, BATCH, 3, 64, SK);
  chan_stats<<<64, 256, 0, stream>>>(c1, mean, rstd, BATCH, 64, SK);
  bn_act<<<blocks(BATCH * 64 * SK), 256, 0, stream>>>(c1, mean, rstd, cg1, cb1, 64, SK, BATCH * 64 * SK, 0.2f);
  conv1x1<<<blocks(BATCH * 256 * SK), 256, 0, stream>>>(c1, cw2, nullptr, c2, BATCH, 64, 256, SK);
  chan_stats<<<256, 256, 0, stream>>>(c2, mean, rstd, BATCH, 256, SK);
  bn_act<<<blocks(BATCH * 256 * SK), 256, 0, stream>>>(c2, mean, rstd, cg2, cb2, 256, SK, BATCH * 256 * SK, 0.2f);
  conv1x1<<<blocks(BATCH * 512 * SK), 256, 0, stream>>>(c2, cw3, nullptr, c3, BATCH, 256, 512, SK);
  chan_stats<<<512, 256, 0, stream>>>(c3, mean, rstd, BATCH, 512, SK);
  bn_act<<<blocks(BATCH * 512 * SK), 256, 0, stream>>>(c3, mean, rstd, cg3, cb3, 512, SK, BATCH * 512 * SK, 0.2f);

  // max over neighborhood -> (B, 512, 16)
  max_over_k<<<blocks(BATCH * 512 * SQ), 256, 0, stream>>>(c3, out);
}
